// HeteroEncoder_180388627077
// MI455X (gfx1250) — compile-verified, matrix operand order checked by reading, not run on hardware
//
#include <hip/hip_runtime.h>
#include <hip/hip_bf16.h>

// ---------------- problem constants ----------------
#define N_TOK 65536
#define D_DIM 512
#define H_DIM 2048
#define T_EXP 8
#define M_TILE 32
#define TILES_MAX (N_TOK / M_TILE)   // 2048

typedef __attribute__((ext_vector_type(16))) _Float16 v16h;
typedef __attribute__((ext_vector_type(8)))  _Float16 half8;
typedef __attribute__((ext_vector_type(8)))  float    v8f;

// Load a 16-lane-striped WMMA A/B fragment for v_wmma_f32_16x16x32_f16.
// Per ISA (CDNA5 ISA §7.12.2, 16-bit A 16x32):
//   lanes 0-15 : elems 0..7 = K 0..7,  elems 8..15 = K 16..23
//   lanes 16-31: elems 0..7 = K 8..15, elems 8..15 = K 24..31
// Caller passes p = &row_or_col[k0 + kb] where kb = (lane>>4)*8; we load
// [p, p+8) and [p+16, p+24) -> two b128 loads (LDS or global).
__device__ __forceinline__ v16h load_frag(const _Float16* __restrict__ p) {
    half8 lo = *(const half8*)(p);
    half8 hi = *(const half8*)(p + 16);
    v16h r;
#pragma unroll
    for (int e = 0; e < 8; ++e) { r[e] = lo[e]; r[8 + e] = hi[e]; }
    return r;
}

// ---------------- routing ----------------
__global__ void zero_cnt_kernel(int* cnt) {
    if (threadIdx.x < T_EXP) cnt[threadIdx.x] = 0;
}

__global__ __launch_bounds__(256) void route_kernel(const int* __restrict__ type_vec,
                                                    int* __restrict__ cnt,
                                                    int* __restrict__ order) {
    int i = blockIdx.x * 256 + threadIdx.x;
    int t = type_vec[i];
    int pos = atomicAdd(&cnt[t], 1);
    order[(size_t)t * N_TOK + pos] = i;
}

// ---------------- LayerNorm 1 (per token row): h_f32 + h_f16 ----------------
__global__ __launch_bounds__(256) void ln1_kernel(const float* __restrict__ x,
                                                  const int* __restrict__ type_vec,
                                                  const float* __restrict__ g1,
                                                  const float* __restrict__ be1,
                                                  float* __restrict__ hf32,
                                                  _Float16* __restrict__ hf16) {
    int row = blockIdx.x;
    int tid = threadIdx.x;
    int t = type_vec[row];
    const float* xr = x + (size_t)row * D_DIM;
    float v0 = xr[tid];
    float v1 = xr[tid + 256];
    float s = v0 + v1, s2 = v0 * v0 + v1 * v1;
#pragma unroll
    for (int off = 16; off >= 1; off >>= 1) {
        s  += __shfl_xor(s,  off, 32);
        s2 += __shfl_xor(s2, off, 32);
    }
    __shared__ float red[16];
    int wave = tid >> 5, lane = tid & 31;
    if (lane == 0) { red[wave] = s; red[8 + wave] = s2; }
    __syncthreads();
    if (wave == 0) {
        float a = (lane < 8) ? red[lane] : 0.f;
        float b = (lane < 8) ? red[8 + lane] : 0.f;
#pragma unroll
        for (int off = 4; off >= 1; off >>= 1) {
            a += __shfl_xor(a, off, 32);
            b += __shfl_xor(b, off, 32);
        }
        if (lane == 0) { red[0] = a; red[1] = b; }
    }
    __syncthreads();
    float mu   = red[0] * (1.0f / D_DIM);
    float var  = red[1] * (1.0f / D_DIM) - mu * mu;
    float rstd = rsqrtf(var + 1e-5f);
    const float* g  = g1  + (size_t)t * D_DIM;
    const float* be = be1 + (size_t)t * D_DIM;
    float h0 = (v0 - mu) * rstd * g[tid]       + be[tid];
    float h1 = (v1 - mu) * rstd * g[tid + 256] + be[tid + 256];
    size_t base = (size_t)row * D_DIM;
    hf32[base + tid]       = h0;
    hf32[base + tid + 256] = h1;
    hf16[base + tid]       = (_Float16)h0;
    hf16[base + tid + 256] = (_Float16)h1;
}

// ---------------- transpose+convert weights: in[R,C] f32 -> out[C,R] f16 ---
__global__ void transpose_cvt_kernel(const float* __restrict__ in,
                                     _Float16* __restrict__ out,
                                     int R, int C) {
    __shared__ float tile[32][33];
    int mat = blockIdx.z;
    const float* src = in + (size_t)mat * R * C;
    _Float16* dst = out + (size_t)mat * R * C;
    int c0 = blockIdx.x * 32, r0 = blockIdx.y * 32;
    int tx = threadIdx.x, ty = threadIdx.y;   // 32 x 8
#pragma unroll
    for (int i = 0; i < 32; i += 8)
        tile[ty + i][tx] = src[(size_t)(r0 + ty + i) * C + c0 + tx];
    __syncthreads();
#pragma unroll
    for (int i = 0; i < 32; i += 8)
        dst[(size_t)(c0 + ty + i) * R + r0 + tx] = (_Float16)tile[tx][ty + i];
}

// ---------------- fused FFN + residual + LN2 per 32-token same-type tile ---
__global__ __launch_bounds__(256)
void ffn_fused_kernel(const _Float16* __restrict__ hf16,
                      const float* __restrict__ hf32,
                      const _Float16* __restrict__ W1t,  // [T][H][D] f16 (col-major K)
                      const _Float16* __restrict__ W2t,  // [T][D][H] f16 (col-major K)
                      const float* __restrict__ b1,      // [T][H]
                      const float* __restrict__ b2,      // [T][D]
                      const float* __restrict__ g2,      // [T][D]
                      const float* __restrict__ be2,     // [T][D]
                      const int* __restrict__ order,     // [T][N]
                      const int* __restrict__ cnt,       // [T]
                      float* __restrict__ out) {         // [N][D]
    extern __shared__ char smem[];
    _Float16* A_lds = (_Float16*)smem;                                  // 32x512  (32 KB)
    _Float16* ACT   = (_Float16*)(smem + M_TILE * D_DIM * 2);           // 32x2048 (128 KB)
    float*    Yb    = (float*)(smem + M_TILE * D_DIM * 2
                                    + M_TILE * H_DIM * 2);              // 32x512  (64 KB)
    __shared__ int idx_s[M_TILE];

    int t    = blockIdx.x / TILES_MAX;
    int tile = blockIdx.x % TILES_MAX;
    int count = cnt[t];
    int j0 = tile * M_TILE;
    if (j0 >= count) return;
    int rows = min(M_TILE, count - j0);

    int tid  = threadIdx.x;
    int lane = tid & 31;
    int wave = tid >> 5;

    if (tid < M_TILE)
        idx_s[tid] = (tid < rows) ? order[(size_t)t * N_TOK + j0 + tid] : -1;
    __syncthreads();

    // ---- stage gathered A rows (f16) into LDS; zero-pad missing rows ----
    {
        int r  = tid >> 3;            // 32 rows, 8 threads per row
        int c0 = (tid & 7) * 64;      // 64 halves each
        _Float16* dst = A_lds + (size_t)r * D_DIM + c0;
        int g = idx_s[r];
        if (g >= 0) {
            const _Float16* src = hf16 + (size_t)g * D_DIM + c0;
#pragma unroll
            for (int u = 0; u < 8; ++u)
                *(half8*)(dst + u * 8) = *(const half8*)(src + u * 8);
        } else {
            half8 z = {};
#pragma unroll
            for (int u = 0; u < 8; ++u)
                *(half8*)(dst + u * 8) = z;
        }
    }
    __syncthreads();

    int rA = lane & 15;              // A row within m-block / B+C column within n-tile
    int kb = (lane >> 4) * 8;        // per-lane K base
    int mh = (lane >> 4) * 8;        // C-layout row base for this lane half

    // ---- GEMM1: ACT[32,2048] = relu(A @ W1[t] + b1[t]) ----
    {
        const _Float16* Wb = W1t + (size_t)t * H_DIM * D_DIM;
        const _Float16* a0 = A_lds + (size_t)rA * D_DIM + kb;
        const _Float16* a1 = A_lds + (size_t)(rA + 16) * D_DIM + kb;
        for (int jn = 0; jn < H_DIM / 16 / 8; ++jn) {   // 16 n-tiles per wave
            int n = (wave + jn * 8) * 16 + rA;          // this lane's column
            const _Float16* bc = Wb + (size_t)n * D_DIM + kb;
            v8f acc0 = {}, acc1 = {};
#pragma unroll 4
            for (int k0 = 0; k0 < D_DIM; k0 += 32) {
                v16h bf  = load_frag(bc + k0);
                v16h af0 = load_frag(a0 + k0);
                v16h af1 = load_frag(a1 + k0);
                acc0 = __builtin_amdgcn_wmma_f32_16x16x32_f16(false, af0, false, bf,
                                                              (short)0, acc0, false, false);
                acc1 = __builtin_amdgcn_wmma_f32_16x16x32_f16(false, af1, false, bf,
                                                              (short)0, acc1, false, false);
            }
            float bias = b1[(size_t)t * H_DIM + n];
#pragma unroll
            for (int v = 0; v < 8; ++v) {
                int m = v + mh;
                float x0 = acc0[v] + bias; x0 = x0 > 0.f ? x0 : 0.f;
                float x1 = acc1[v] + bias; x1 = x1 > 0.f ? x1 : 0.f;
                ACT[(size_t)m * H_DIM + n]        = (_Float16)x0;
                ACT[(size_t)(m + 16) * H_DIM + n] = (_Float16)x1;
            }
        }
    }
    __syncthreads();

    // ---- GEMM2: Y[32,512] = ACT @ W2[t] + b2[t] + h (residual) ----
    {
        const _Float16* Wb = W2t + (size_t)t * D_DIM * H_DIM;
        const _Float16* a0 = ACT + (size_t)rA * H_DIM + kb;
        const _Float16* a1 = ACT + (size_t)(rA + 16) * H_DIM + kb;
        for (int jn = 0; jn < D_DIM / 16 / 8; ++jn) {   // 4 n-tiles per wave
            int n = (wave + jn * 8) * 16 + rA;
            const _Float16* bc = Wb + (size_t)n * H_DIM + kb;
            v8f acc0 = {}, acc1 = {};
#pragma unroll 4
            for (int k0 = 0; k0 < H_DIM; k0 += 32) {
                v16h bf  = load_frag(bc + k0);
                v16h af0 = load_frag(a0 + k0);
                v16h af1 = load_frag(a1 + k0);
                acc0 = __builtin_amdgcn_wmma_f32_16x16x32_f16(false, af0, false, bf,
                                                              (short)0, acc0, false, false);
                acc1 = __builtin_amdgcn_wmma_f32_16x16x32_f16(false, af1, false, bf,
                                                              (short)0, acc1, false, false);
            }
            float bias = b2[(size_t)t * D_DIM + n];
#pragma unroll
            for (int v = 0; v < 8; ++v) {
                int m = v + mh;
                int ga = idx_s[m], gb = idx_s[m + 16];
                float r0 = (ga >= 0) ? hf32[(size_t)ga * D_DIM + n] : 0.f;
                float r1 = (gb >= 0) ? hf32[(size_t)gb * D_DIM + n] : 0.f;
                Yb[(size_t)m * D_DIM + n]        = acc0[v] + bias + r0;
                Yb[(size_t)(m + 16) * D_DIM + n] = acc1[v] + bias + r1;
            }
        }
    }
    __syncthreads();

    // ---- LN2 per row (one wave handles 4 rows) + writeout ----
    for (int r = wave; r < M_TILE; r += 8) {
        int g = idx_s[r];
        if (g < 0) continue;
        float vals[16];
        float s = 0.f, s2 = 0.f;
#pragma unroll
        for (int u = 0; u < 16; ++u) {
            float v = Yb[(size_t)r * D_DIM + u * 32 + lane];
            vals[u] = v; s += v; s2 += v * v;
        }
#pragma unroll
        for (int off = 16; off >= 1; off >>= 1) {
            s  += __shfl_xor(s,  off, 32);
            s2 += __shfl_xor(s2, off, 32);
        }
        float mu   = s * (1.0f / D_DIM);
        float var  = s2 * (1.0f / D_DIM) - mu * mu;
        float rstd = rsqrtf(var + 1e-5f);
#pragma unroll
        for (int u = 0; u < 16; ++u) {
            int d = u * 32 + lane;
            out[(size_t)g * D_DIM + d] =
                (vals[u] - mu) * rstd * g2[(size_t)t * D_DIM + d]
                + be2[(size_t)t * D_DIM + d];
        }
    }
}

// ---------------- host launch ----------------
extern "C" void kernel_launch(void* const* d_in, const int* in_sizes, int n_in,
                              void* d_out, int out_size, void* d_ws, size_t ws_size,
                              hipStream_t stream) {
    const float* x        = (const float*)d_in[0];
    const int*   type_vec = (const int*)  d_in[1];
    const float* W1       = (const float*)d_in[2];
    const float* b1       = (const float*)d_in[3];
    const float* W2       = (const float*)d_in[4];
    const float* b2       = (const float*)d_in[5];
    const float* g1       = (const float*)d_in[6];
    const float* be1      = (const float*)d_in[7];
    const float* g2       = (const float*)d_in[8];
    const float* be2      = (const float*)d_in[9];
    float* out = (float*)d_out;

    // workspace layout
    char* ws = (char*)d_ws;
    size_t off = 0;
    int* cnt = (int*)(ws + off);                 off += 256;
    int* order = (int*)(ws + off);               off += (size_t)T_EXP * N_TOK * 4;
    float* hf32 = (float*)(ws + off);            off += (size_t)N_TOK * D_DIM * 4;
    _Float16* hf16 = (_Float16*)(ws + off);      off += (size_t)N_TOK * D_DIM * 2;
    _Float16* W1t = (_Float16*)(ws + off);       off += (size_t)T_EXP * D_DIM * H_DIM * 2;
    _Float16* W2t = (_Float16*)(ws + off);       off += (size_t)T_EXP * D_DIM * H_DIM * 2;
    (void)ws_size; (void)n_in; (void)in_sizes; (void)out_size;

    zero_cnt_kernel<<<1, 32, 0, stream>>>(cnt);
    route_kernel<<<N_TOK / 256, 256, 0, stream>>>(type_vec, cnt, order);
    ln1_kernel<<<N_TOK, 256, 0, stream>>>(x, type_vec, g1, be1, hf32, hf16);

    // W1: [T][D][H] f32 -> W1t [T][H][D] f16   (R=D, C=H)
    transpose_cvt_kernel<<<dim3(H_DIM / 32, D_DIM / 32, T_EXP), dim3(32, 8), 0, stream>>>(
        W1, W1t, D_DIM, H_DIM);
    // W2: [T][H][D] f32 -> W2t [T][D][H] f16   (R=H, C=D)
    transpose_cvt_kernel<<<dim3(D_DIM / 32, H_DIM / 32, T_EXP), dim3(32, 8), 0, stream>>>(
        W2, W2t, H_DIM, D_DIM);

    const int smem_bytes = M_TILE * D_DIM * 2     // A tile f16  (32 KB)
                         + M_TILE * H_DIM * 2     // activations (128 KB)
                         + M_TILE * D_DIM * 4;    // residual f32 (64 KB) = 224 KB
    hipFuncSetAttribute((const void*)ffn_fused_kernel,
                        hipFuncAttributeMaxDynamicSharedMemorySize, smem_bytes);
    ffn_fused_kernel<<<T_EXP * TILES_MAX, 256, smem_bytes, stream>>>(
        hf16, hf32, W1t, W2t, b1, b2, g2, be2, order, cnt, out);
}
